// Decoder_88072599372130
// MI455X (gfx1250) — matrix operand order repaired and checked by reading, not compiled, not run on hardware
//
#include <hip/hip_runtime.h>
#include <hip/hip_bf16.h>

// ---------------------------------------------------------------------------
// Transformer decoder layer for MI455X (gfx1250, wave32, WMMA).
// All GEMMs run through v_wmma_f32_16x16x32_bf16 (f32 accumulate).
// Compute-bound (~410 GFLOP vs ~500MB HBM traffic @23.3TB/s). Weights
// (~34MB bf16) live entirely in the 192MB L2, so GEMMs stream fragments
// straight from global/L2 with b128 clauses; wave tile 32x64 gives 8 WMMAs
// per 6 fragment loads (~21 flop per L2 byte). V is pre-transposed once
// (staged via global_load_async_to_lds_b128 / ASYNCcnt) so every GEMM uses
// contiguous K-major fragment loads. Causal score blocks fully above the
// diagonal skip their K loop. A-operand gets a WGP-scope prefetch stream.
// ---------------------------------------------------------------------------

typedef __attribute__((ext_vector_type(16))) __bf16 v16bf;
typedef __attribute__((ext_vector_type(8)))  __bf16 v8bf;
typedef __attribute__((ext_vector_type(8)))  float  v8f;

#define EDIM 1024
#define FDIM 4096
#define BDIM 4
#define SDIM 2048

__device__ __forceinline__ v8f wmma_bf16(v16bf a, v16bf b, v8f c) {
  // D = A(16x32 bf16) x B(32x16 bf16) + C(16x16 f32)
  return __builtin_amdgcn_wmma_f32_16x16x32_bf16(
      /*neg_a=*/false, a, /*neg_b=*/false, b,
      /*c_mod=*/(short)0, c, /*reuse_a=*/false, /*reuse_b=*/false);
}

// A-fragment (16x32, K-contiguous rows). contiguous k order, same as frag_bt:
// lanes 0-15: row M=lane, K = k..k+15
// lanes 16-31: row M=lane-16, K = k+16..k+31
__device__ __forceinline__ v16bf frag_a(const __bf16* p, int ld, int k, int lane) {
  const __bf16* q = p + (size_t)(lane & 15) * ld + (k + ((lane & 16) ? 16 : 0));
  v16bf f;
  ((v8bf*)&f)[0] = *(const v8bf*)q;         // 16B aligned
  ((v8bf*)&f)[1] = *(const v8bf*)(q + 8);
  return f;
}

// B-fragment from W stored (N,K) row-major (B = W^T, column n = row n of W).
// lanes 0-15: col N=lane, K = k..k+15 ; lanes 16-31: col N=lane-16, K=k+16..k+31
__device__ __forceinline__ v16bf frag_bt(const __bf16* p, int ld, int k, int lane) {
  const __bf16* q = p + (size_t)(lane & 15) * ld + (k + ((lane & 16) ? 16 : 0));
  return *(const v16bf*)q;                  // 32B aligned contiguous
}

// ---------------------------------------------------------------------------
// WMMA GEMM:  C[z] = act( scale * A[z] @ B[z]^T + bias + res )
//   A: (M,K) bf16.  Bm: (N,K) bf16 (K contiguous).
// Block: 256 threads (8 waves). Block tile 64x256; wave tile 32x64 (2x4 WMMA).
// ---------------------------------------------------------------------------
template <bool CAUSAL, bool RELU, bool HASBIAS, bool HASRES, bool OUTF, bool OUTB>
__global__ __launch_bounds__(256) void gemm_wmma(
    const __bf16* __restrict__ A, const __bf16* __restrict__ Bm,
    const float* __restrict__ bias, const float* __restrict__ res,
    float* __restrict__ Cf, __bf16* __restrict__ Cb,
    int M, int N, int K, float scale,
    long long sA, long long sB, long long sC) {
  const int lane = threadIdx.x & 31;
  const int wave = threadIdx.x >> 5;
  const int row0 = blockIdx.x * 64 + (wave & 1) * 32;
  const int col0 = blockIdx.y * 256 + (wave >> 1) * 64;
  const long long z = blockIdx.z;

  // Causal early-out: whole block strictly above the diagonal -> write -inf.
  if (CAUSAL && OUTF) {
    if ((int)blockIdx.y * 256 > (int)blockIdx.x * 64 + 63) {
      const int nl = lane & 15;
      const int mo = (lane & 16) ? 8 : 0;
#pragma unroll
      for (int i = 0; i < 2; ++i)
#pragma unroll
        for (int j = 0; j < 4; ++j)
#pragma unroll
          for (int v = 0; v < 8; ++v) {
            const int m = row0 + 16 * i + mo + v;
            const int n = col0 + 16 * j + nl;
            Cf[(size_t)z * (size_t)sC + (size_t)m * N + n] = -1e30f;
          }
      return;
    }
  }

  const __bf16* Ab = A + z * sA;
  const __bf16* Bb = Bm + z * sB;

  v8f acc[2][4] = {};

  for (int k = 0; k < K; k += 32) {
    // Stream-ahead prefetch of the A operand (activations come from HBM on
    // first touch; weights/B are L2-resident). Locality 3 -> WGP scope, pulls
    // the line into all cache levels. ~1KB ahead in this lane's row.
    __builtin_prefetch(Ab + (size_t)(row0 + (lane & 15)) * K + k + 512, 0, 3);
    v16bf a[2], b[4];
#pragma unroll
    for (int i = 0; i < 2; ++i)
      a[i] = frag_a(Ab + (size_t)(row0 + 16 * i) * K, K, k, lane);
#pragma unroll
    for (int j = 0; j < 4; ++j)
      b[j] = frag_bt(Bb + (size_t)(col0 + 16 * j) * K, K, k, lane);
#pragma unroll
    for (int i = 0; i < 2; ++i)
#pragma unroll
      for (int j = 0; j < 4; ++j)
        acc[i][j] = wmma_bf16(a[i], b[j], acc[i][j]);
  }

  // Epilogue. C/D layout (ISA 7.12.2): VGPR v, lanes 0-15 -> M=r0+v, N=c0+lane;
  // lanes 16-31 -> M=r0+8+v, N=c0+lane-16.
  auto epi = [&](v8f acc1, int r0, int c0) {
    const int n = c0 + (lane & 15);
    const int m0 = r0 + ((lane & 16) ? 8 : 0);
    float bn = 0.0f;
    if (HASBIAS) bn = bias[n];
#pragma unroll
    for (int v = 0; v < 8; ++v) {
      const int m = m0 + v;
      float x = acc1[v] * scale + bn;
      const size_t idx = (size_t)z * (size_t)sC + (size_t)m * N + n;
      if (HASRES) x += res[idx];
      if (CAUSAL) { if (n > m) x = -1e30f; }
      if (RELU) x = fmaxf(x, 0.0f);
      if (OUTF) Cf[idx] = x;
      if (OUTB) Cb[idx] = (__bf16)x;
    }
  };
#pragma unroll
  for (int i = 0; i < 2; ++i)
#pragma unroll
    for (int j = 0; j < 4; ++j)
      epi(acc[i][j], row0 + 16 * i, col0 + 16 * j);
}

// ---------------------------------------------------------------------------
// Batched transpose (R,C) -> (C,R) bf16. 64x32 tiles are staged into LDS with
// async global->LDS DMA (gfx1250 GLOBAL_LOAD_ASYNC_TO_LDS_B128, ASYNCcnt),
// then written out transposed with contiguous stores. LDS row stride padded
// to 68B (17 DWORDs, coprime with 64 banks) for conflict-free column reads.
// ---------------------------------------------------------------------------
__global__ __launch_bounds__(256) void transpose_bf16(
    const __bf16* __restrict__ in, __bf16* __restrict__ out, int R, int C,
    long long sIn, long long sOut) {
  __shared__ __bf16 t[64][34];
  const long long z = blockIdx.z;
  const __bf16* ib = in + z * sIn;
  __bf16* ob = out + z * sOut;
  const int r0 = blockIdx.x * 64;  // over R
  const int c0 = blockIdx.y * 32;  // over C

  // Async stage: each of 256 lanes DMA-copies 16B (8 bf16): row = tid/4,
  // 8-elem segment = tid%4.
  {
    const int row = threadIdx.x >> 2;
    const int seg = (threadIdx.x & 3) * 8;
    const __bf16* gsrc = ib + (size_t)(r0 + row) * C + c0 + seg;
    unsigned lds_addr = (unsigned)(uintptr_t)(&t[row][seg]);  // addr[31:0] = LDS offset
    asm volatile("global_load_async_to_lds_b128 %0, %1, off"
                 :
                 : "v"(lds_addr), "v"(gsrc)
                 : "memory");
  }
  asm volatile("s_wait_asynccnt 0x0" ::: "memory");
  __syncthreads();

  // Transposed write-out: thread group per output row c; 64 contiguous r's.
  const int tr = threadIdx.x & 63;        // r offset within tile
  const int cg = (threadIdx.x >> 6) * 8;  // 4 groups x 8 columns
#pragma unroll
  for (int j = 0; j < 8; ++j) {
    const int c = cg + j;
    ob[(size_t)(c0 + c) * R + r0 + tr] = t[tr][c];
  }
}

// ---------------------------------------------------------------------------
// LayerNorm: one block per row of E elements; outputs bf16 (GEMM input).
// ---------------------------------------------------------------------------
__global__ __launch_bounds__(256) void ln_kernel(
    const float* __restrict__ x, const float* __restrict__ g,
    const float* __restrict__ b, __bf16* __restrict__ y, int E) {
  const size_t row = blockIdx.x;
  const float* xr = x + row * E;
  float s = 0.0f, s2 = 0.0f;
  for (int i = threadIdx.x; i < E; i += blockDim.x) {
    float v = xr[i];
    s += v;
    s2 += v * v;
  }
#pragma unroll
  for (int off = 16; off; off >>= 1) {
    s += __shfl_xor(s, off, 32);
    s2 += __shfl_xor(s2, off, 32);
  }
  __shared__ float rs[8], rs2[8];
  const int wave = threadIdx.x >> 5, lane = threadIdx.x & 31;
  if (lane == 0) { rs[wave] = s; rs2[wave] = s2; }
  __syncthreads();
  s = 0.0f; s2 = 0.0f;
#pragma unroll
  for (int w = 0; w < 8; ++w) { s += rs[w]; s2 += rs2[w]; }
  const float mu = s / E;
  const float var = s2 / E - mu * mu;
  const float inv = rsqrtf(var + 1e-5f);
  for (int i = threadIdx.x; i < E; i += blockDim.x) {
    float v = (xr[i] - mu) * inv * g[i] + b[i];
    y[row * E + i] = (__bf16)v;
  }
}

// ---------------------------------------------------------------------------
// Row softmax over S columns: scores f32 -> probs bf16. Block per (q, batch).
// ---------------------------------------------------------------------------
__global__ __launch_bounds__(256) void softmax_kernel(
    const float* __restrict__ sc, __bf16* __restrict__ P, int S) {
  const size_t base = ((size_t)blockIdx.y * S + blockIdx.x) * S;
  const float* r = sc + base;
  const int wave = threadIdx.x >> 5, lane = threadIdx.x & 31;

  float mx = -3.4e38f;
  for (int i = threadIdx.x; i < S; i += blockDim.x) mx = fmaxf(mx, r[i]);
#pragma unroll
  for (int off = 16; off; off >>= 1) mx = fmaxf(mx, __shfl_xor(mx, off, 32));
  __shared__ float wmx[8];
  if (lane == 0) wmx[wave] = mx;
  __syncthreads();
  mx = wmx[0];
#pragma unroll
  for (int w = 1; w < 8; ++w) mx = fmaxf(mx, wmx[w]);

  float sum = 0.0f;
  for (int i = threadIdx.x; i < S; i += blockDim.x) sum += __expf(r[i] - mx);
#pragma unroll
  for (int off = 16; off; off >>= 1) sum += __shfl_xor(sum, off, 32);
  __shared__ float wsm[8];
  if (lane == 0) wsm[wave] = sum;
  __syncthreads();
  sum = 0.0f;
#pragma unroll
  for (int w = 0; w < 8; ++w) sum += wsm[w];
  const float inv = 1.0f / sum;
  for (int i = threadIdx.x; i < S; i += blockDim.x)
    P[base + i] = (__bf16)(__expf(r[i] - mx) * inv);
}

// f32 -> bf16 bulk conversion (weights / src_encs)
__global__ __launch_bounds__(256) void cvt_kernel(const float* __restrict__ in,
                                                  __bf16* __restrict__ out,
                                                  size_t n) {
  size_t i = (size_t)blockIdx.x * blockDim.x + threadIdx.x;
  if (i < n) out[i] = (__bf16)in[i];
}

// ---------------------------------------------------------------------------
extern "C" void kernel_launch(void* const* d_in, const int* in_sizes, int n_in,
                              void* d_out, int out_size, void* d_ws, size_t ws_size,
                              hipStream_t stream) {
  (void)in_sizes; (void)n_in; (void)out_size; (void)ws_size;
  const int E = EDIM, F = FDIM, Bn = BDIM, S = SDIM;
  const int M = Bn * S;  // 8192 rows across batch

  const float* src   = (const float*)d_in[0];
  const float* tgt   = (const float*)d_in[1];
  const float* ln1_g = (const float*)d_in[2];
  const float* ln1_b = (const float*)d_in[3];
  const float* ln2_g = (const float*)d_in[4];
  const float* ln2_b = (const float*)d_in[5];
  const float* ln3_g = (const float*)d_in[6];
  const float* ln3_b = (const float*)d_in[7];
  const float* Wf32[10] = {
      (const float*)d_in[8],  (const float*)d_in[10], (const float*)d_in[12],
      (const float*)d_in[14], (const float*)d_in[16], (const float*)d_in[18],
      (const float*)d_in[20], (const float*)d_in[22], (const float*)d_in[24],
      (const float*)d_in[26]};
  const float* bia[10] = {
      (const float*)d_in[9],  (const float*)d_in[11], (const float*)d_in[13],
      (const float*)d_in[15], (const float*)d_in[17], (const float*)d_in[19],
      (const float*)d_in[21], (const float*)d_in[23], (const float*)d_in[25],
      (const float*)d_in[27]};
  float* out = (float*)d_out;

  // ---- workspace carve-up (256B aligned) ----
  char* wsp = (char*)d_ws;
  auto alloc = [&](size_t bytes) -> void* {
    void* p = (void*)wsp;
    wsp += (bytes + 255) & ~(size_t)255;
    return p;
  };
  const size_t wElems[10] = {(size_t)E * E, (size_t)E * E, (size_t)E * E,
                             (size_t)E * E, (size_t)E * E, (size_t)E * E,
                             (size_t)E * E, (size_t)E * E, (size_t)F * E,
                             (size_t)E * F};
  __bf16* W16[10];
  for (int i = 0; i < 10; ++i) W16[i] = (__bf16*)alloc(wElems[i] * 2);
  __bf16* src16 = (__bf16*)alloc((size_t)M * E * 2);
  __bf16* xn16  = (__bf16*)alloc((size_t)M * E * 2);
  __bf16* Q16   = (__bf16*)alloc((size_t)M * E * 2);
  __bf16* K16   = (__bf16*)alloc((size_t)M * E * 2);
  __bf16* V16   = (__bf16*)alloc((size_t)M * E * 2);
  __bf16* Vt16  = (__bf16*)alloc((size_t)M * E * 2);  // per-batch (E,S)
  __bf16* ao16  = (__bf16*)alloc((size_t)M * E * 2);
  __bf16* h16   = (__bf16*)alloc((size_t)M * F * 2);
  float*  attnf = (float*)alloc((size_t)M * E * 4);
  float*  crossf= (float*)alloc((size_t)M * E * 4);
  float*  scbuf = (float*)alloc((size_t)Bn * S * S * 4);
  __bf16* P16   = (__bf16*)alloc((size_t)Bn * S * S * 2);

  const long long sSE = (long long)S * E;   // per-batch activation stride
  const long long sSS = (long long)S * S;   // per-batch score stride
  const float kScale = 1.0f / 32.0f;        // 1/sqrt(E), E=1024

  // ---- convert weights + src_encs to bf16 ----
  for (int i = 0; i < 10; ++i) {
    size_t n = wElems[i];
    cvt_kernel<<<dim3((unsigned)((n + 255) / 256)), 256, 0, stream>>>(Wf32[i], W16[i], n);
  }
  {
    size_t n = (size_t)M * E;
    cvt_kernel<<<dim3((unsigned)((n + 255) / 256)), 256, 0, stream>>>(src, src16, n);
  }

  const dim3 blk(256);
  const dim3 gE(M / 64, E / 256, 1);        // (128, 4)
  const dim3 gF(M / 64, F / 256, 1);        // (128, 16)
  const dim3 gSc(S / 64, S / 256, Bn);      // (32, 8, 4)
  const dim3 gAV(S / 64, E / 256, Bn);      // (32, 4, 4)
  const dim3 gT(S / 64, E / 32, Bn);        // V transpose tiles (64x32)

  // ======================= self-attention =======================
  ln_kernel<<<dim3(M), blk, 0, stream>>>(tgt, ln1_g, ln1_b, xn16, E);

  gemm_wmma<false, false, true, false, false, true><<<gE, blk, 0, stream>>>(
      xn16, W16[0], bia[0], nullptr, nullptr, Q16, M, E, E, 1.0f, 0, 0, 0);
  gemm_wmma<false, false, true, false, false, true><<<gE, blk, 0, stream>>>(
      xn16, W16[1], bia[1], nullptr, nullptr, K16, M, E, E, 1.0f, 0, 0, 0);
  gemm_wmma<false, false, true, false, false, true><<<gE, blk, 0, stream>>>(
      xn16, W16[2], bia[2], nullptr, nullptr, V16, M, E, E, 1.0f, 0, 0, 0);
  transpose_bf16<<<gT, blk, 0, stream>>>(V16, Vt16, S, E, sSE, sSE);

  // scores = Q K^T / 32 with causal mask (f32 out)
  gemm_wmma<true, false, false, false, true, false><<<gSc, blk, 0, stream>>>(
      Q16, K16, nullptr, nullptr, scbuf, nullptr, S, S, E, kScale, sSE, sSE, sSS);
  softmax_kernel<<<dim3(S, Bn), blk, 0, stream>>>(scbuf, P16, S);
  // attn_out = P @ V = P @ Vt^T (bf16 out)
  gemm_wmma<false, false, false, false, false, true><<<gAV, blk, 0, stream>>>(
      P16, Vt16, nullptr, nullptr, nullptr, ao16, S, E, S, 1.0f, sSS, sSE, sSE);
  // out proj + residual with tgt  -> attnf (f32)
  gemm_wmma<false, false, true, true, true, false><<<gE, blk, 0, stream>>>(
      ao16, W16[3], bia[3], tgt, attnf, nullptr, M, E, E, 1.0f, 0, 0, 0);

  // ======================= cross-attention =======================
  ln_kernel<<<dim3(M), blk, 0, stream>>>(attnf, ln2_g, ln2_b, xn16, E);
  gemm_wmma<false, false, true, false, false, true><<<gE, blk, 0, stream>>>(
      xn16, W16[4], bia[4], nullptr, nullptr, Q16, M, E, E, 1.0f, 0, 0, 0);
  gemm_wmma<false, false, true, false, false, true><<<gE, blk, 0, stream>>>(
      src16, W16[5], bia[5], nullptr, nullptr, K16, M, E, E, 1.0f, 0, 0, 0);
  gemm_wmma<false, false, true, false, false, true><<<gE, blk, 0, stream>>>(
      src16, W16[6], bia[6], nullptr, nullptr, V16, M, E, E, 1.0f, 0, 0, 0);
  transpose_bf16<<<gT, blk, 0, stream>>>(V16, Vt16, S, E, sSE, sSE);

  gemm_wmma<false, false, false, false, true, false><<<gSc, blk, 0, stream>>>(
      Q16, K16, nullptr, nullptr, scbuf, nullptr, S, S, E, kScale, sSE, sSE, sSS);
  softmax_kernel<<<dim3(S, Bn), blk, 0, stream>>>(scbuf, P16, S);
  gemm_wmma<false, false, false, false, false, true><<<gAV, blk, 0, stream>>>(
      P16, Vt16, nullptr, nullptr, nullptr, ao16, S, E, S, 1.0f, sSS, sSE, sSE);
  gemm_wmma<false, false, true, true, true, false><<<gE, blk, 0, stream>>>(
      ao16, W16[7], bia[7], attnf, crossf, nullptr, M, E, E, 1.0f, 0, 0, 0);

  // ======================= feed-forward =======================
  ln_kernel<<<dim3(M), blk, 0, stream>>>(crossf, ln3_g, ln3_b, xn16, E);
  gemm_wmma<false, true, true, false, false, true><<<gF, blk, 0, stream>>>(
      xn16, W16[8], bia[8], nullptr, nullptr, h16, M, F, E, 1.0f, 0, 0, 0);
  gemm_wmma<false, false, true, true, true, false><<<gE, blk, 0, stream>>>(
      h16, W16[9], bia[9], crossf, out, nullptr, M, E, F, 1.0f, 0, 0, 0);
}
